// GaussianRBF_81750407512455
// MI455X (gfx1250) — matrix operand order repaired and checked by reading, not compiled, not run on hardware
//
#include <hip/hip_runtime.h>

// ---------------- types ----------------
typedef __bf16 bf16_t;
typedef bf16_t v16bf __attribute__((ext_vector_type(16)));
typedef float  v8f   __attribute__((ext_vector_type(8)));
typedef unsigned int u32x4 __attribute__((ext_vector_type(4)));

union FragBF16 {
    u32x4 q[2];   // two 16B chunks
    v16bf f;      // 16 bf16 elements
};

__device__ __forceinline__ unsigned short f2bf(float f) {
    unsigned int u = __float_as_uint(f);
    u += 0x7FFFu + ((u >> 16) & 1u);      // round-to-nearest-even
    return (unsigned short)(u >> 16);
}

// dims fixed by the reference
#define MB 8192   // batch rows
#define KK 1024   // n_a
#define NN 4096   // units

// ---------------- kernel 1: x -> bf16 + row ||x||^2 (fp32 exact) ----------------
__global__ __launch_bounds__(256)
void xprep(const float* __restrict__ x, unsigned short* __restrict__ xb,
           float* __restrict__ xsq) {
    const int row = blockIdx.x;
    const int t   = threadIdx.x;
    const float4 v = ((const float4*)(x + (size_t)row * KK))[t];   // 256*4 = 1024
    float s = v.x * v.x + v.y * v.y + v.z * v.z + v.w * v.w;

    unsigned int lo = (unsigned int)f2bf(v.x) | ((unsigned int)f2bf(v.y) << 16);
    unsigned int hi = (unsigned int)f2bf(v.z) | ((unsigned int)f2bf(v.w) << 16);
    ((uint2*)(xb + (size_t)row * KK))[t] = make_uint2(lo, hi);

    __shared__ float red[256];
    red[t] = s;
    __syncthreads();
    for (int off = 128; off > 0; off >>= 1) {
        if (t < off) red[t] += red[t + off];
        __syncthreads();
    }
    if (t == 0) xsq[row] = red[0];
}

// ---------------- kernel 2: mu (K x N) -> muT bf16 (N x K) ----------------
__global__ __launch_bounds__(256)
void muprep(const float* __restrict__ mu, unsigned short* __restrict__ mtb) {
    __shared__ unsigned short tile[32 * 33];
    const int n0 = blockIdx.x * 32;
    const int k0 = blockIdx.y * 32;
    const int tid = threadIdx.x;
    const int nl = tid & 31;       // column within tile (coalesced read)
    const int kl = tid >> 5;       // 0..7
#pragma unroll
    for (int j = 0; j < 4; ++j) {
        int k = kl + j * 8;
        float v = mu[(size_t)(k0 + k) * NN + n0 + nl];
        tile[nl * 33 + k] = f2bf(v);
    }
    __syncthreads();
    const int row = tid >> 3;          // 0..31 (n within tile)
    const int ko  = (tid & 7) * 4;     // k chunk of 4 u16 = 8B
    union { unsigned short us[4]; unsigned long long u; } p;
    p.us[0] = tile[row * 33 + ko + 0];
    p.us[1] = tile[row * 33 + ko + 1];
    p.us[2] = tile[row * 33 + ko + 2];
    p.us[3] = tile[row * 33 + ko + 3];
    *(unsigned long long*)(mtb + (size_t)(n0 + row) * KK + k0 + ko) = p.u;
}

// ---------------- kernel 3: column sums ||mu_col||^2 (fp32, coalesced) ----------------
__global__ __launch_bounds__(256)
void musqk(const float* __restrict__ mu, float* __restrict__ musq) {
    const int n = blockIdx.x * 256 + threadIdx.x;   // 0..4095
    float s = 0.0f;
    for (int k = 0; k < KK; ++k) {
        float v = mu[(size_t)k * NN + n];
        s = fmaf(v, v, s);
    }
    musq[n] = s;
}

// ---------------- kernel 4: WMMA GEMM + fused RBF epilogue ----------------
// block tile 128(M) x 128(N), K-step 32, 256 threads = 8 waves (2x4 wave grid),
// each wave: 64x32 = 4x2 tiles of 16x16, 8 v_wmma_f32_16x16x32_bf16 per K-step.
// Staging: double-buffered GLOBAL_LOAD_ASYNC_TO_LDS_B128 (ASYNCcnt pipeline).
#define LDS_PITCH 40   // u16 units: 32 data + 8 pad (80B, multiple of 16B)
#define NTILES (KK / 32)

// Issue one 128x32 A tile + one 128x32 B tile as async global->LDS copies.
// 4 instructions per wave (each instruction moves 16B per lane).
__device__ __forceinline__ void issue_async_tile(
    const unsigned short* __restrict__ xb, const unsigned short* __restrict__ mtb,
    unsigned short* AsBuf, unsigned short* BsBuf,
    int m0, int n0, int k0, int tid) {
#pragma unroll
    for (int i = 0; i < 2; ++i) {
        int c   = tid + i * 256;          // 0..511
        int row = c >> 2;                 // 0..127
        int off = (c & 3) << 3;           // 0,8,16,24 (u16 units)
        unsigned la = (unsigned)(size_t)(AsBuf + row * LDS_PITCH + off);
        unsigned long long ga =
            (unsigned long long)(size_t)(xb + (size_t)(m0 + row) * KK + k0 + off);
        asm volatile("global_load_async_to_lds_b128 %0, %1, off"
                     :: "v"(la), "v"(ga) : "memory");
        unsigned lb = (unsigned)(size_t)(BsBuf + row * LDS_PITCH + off);
        unsigned long long gb =
            (unsigned long long)(size_t)(mtb + (size_t)(n0 + row) * KK + k0 + off);
        asm volatile("global_load_async_to_lds_b128 %0, %1, off"
                     :: "v"(lb), "v"(gb) : "memory");
    }
}

__global__ __launch_bounds__(256)
void rbf_gemm(const unsigned short* __restrict__ xb,    // MB x KK bf16
              const unsigned short* __restrict__ mtb,   // NN x KK bf16 (mu^T)
              const float* __restrict__ xsq,
              const float* __restrict__ musq,
              const float* __restrict__ gammap,
              float* __restrict__ out) {
    __shared__ __align__(16) unsigned short As[2][128 * LDS_PITCH];
    __shared__ __align__(16) unsigned short Bs[2][128 * LDS_PITCH];

    const int tid  = threadIdx.x;
    const int lane = tid & 31;
    const int wave = tid >> 5;
    const int wm   = wave & 1;          // 0..1 along M
    const int wn   = wave >> 1;         // 0..3 along N
    const int m0   = blockIdx.y * 128;
    const int n0   = blockIdx.x * 128;
    const int r    = lane & 15;
    const int half = lane >> 4;

    v8f acc[4][2];
    const v8f vzero = {0.f, 0.f, 0.f, 0.f, 0.f, 0.f, 0.f, 0.f};
#pragma unroll
    for (int mt = 0; mt < 4; ++mt)
#pragma unroll
        for (int nt = 0; nt < 2; ++nt) acc[mt][nt] = vzero;

    // prologue: async-load tile 0 into buffer 0
    issue_async_tile(xb, mtb, As[0], Bs[0], m0, n0, 0, tid);

    for (int kt = 0; kt < NTILES; ++kt) {
        const int buf = kt & 1;
        if (kt + 1 < NTILES) {
            // overlap: issue tile kt+1 into the other buffer, then wait only
            // for tile kt's 4 async ops (they retire in order per wave)
            issue_async_tile(xb, mtb, As[buf ^ 1], Bs[buf ^ 1],
                             m0, n0, (kt + 1) * 32, tid);
            asm volatile("s_wait_asynccnt 0x4" ::: "memory");
        } else {
            asm volatile("s_wait_asynccnt 0x0" ::: "memory");
        }
        __syncthreads();   // all waves' tile-kt async writes have landed

        // A fragments: 16x32, lane layout: row = lane%16,
        //   lo 16B = K[half*8 .. half*8+7], hi 16B = K[16+half*8 ..]
        FragBF16 a[4];
#pragma unroll
        for (int mt = 0; mt < 4; ++mt) {
            const unsigned short* ap = As[buf] + (wm * 64 + mt * 16 + r) * LDS_PITCH;
            a[mt].q[0] = *(const u32x4*)(ap + half * 8);
            a[mt].q[1] = *(const u32x4*)(ap + 16 + half * 8);
        }
        // B fragments: 32x16, lane layout: col = lane%16,
        //   lo 16B = K[half*8 .. half*8+7], hi 16B = K[16+half*8 ..]
        FragBF16 b[2];
#pragma unroll
        for (int nt = 0; nt < 2; ++nt) {
            const unsigned short* bp =
                Bs[buf] + (wn * 32 + nt * 16 + r) * LDS_PITCH + half * 8;
            b[nt].q[0] = *(const u32x4*)(bp);
            b[nt].q[1] = *(const u32x4*)(bp + 16);
        }
#pragma unroll
        for (int mt = 0; mt < 4; ++mt)
#pragma unroll
            for (int nt = 0; nt < 2; ++nt)
                acc[mt][nt] = __builtin_amdgcn_wmma_f32_16x16x32_bf16(
                    false, a[mt].f, false, b[nt].f,
                    (short)0, acc[mt][nt], false, false);

        __syncthreads();   // all waves done reading buf before it is overwritten
    }

    // epilogue: out = exp(-g * (||x||^2 + ||mu||^2 - 2*cross))
    const float g = gammap[0];
#pragma unroll
    for (int mt = 0; mt < 4; ++mt) {
        const int gmb = m0 + wm * 64 + mt * 16 + half * 8;   // C layout: M = vgpr + half*8
        float xs[8];
#pragma unroll
        for (int e = 0; e < 8; ++e) xs[e] = xsq[gmb + e];
#pragma unroll
        for (int nt = 0; nt < 2; ++nt) {
            const int gn = n0 + wn * 32 + nt * 16 + r;       // C layout: N = lane%16
            const float ms = musq[gn];
            float* op = out + (size_t)gmb * NN + gn;
#pragma unroll
            for (int e = 0; e < 8; ++e) {
                float l2 = xs[e] + ms - 2.0f * acc[mt][nt][e];
                op[(size_t)e * NN] = __expf(-g * l2);
            }
        }
    }
}

// ---------------- host launcher ----------------
extern "C" void kernel_launch(void* const* d_in, const int* in_sizes, int n_in,
                              void* d_out, int out_size, void* d_ws, size_t ws_size,
                              hipStream_t stream) {
    const float* x     = (const float*)d_in[0];   // 8192 x 1024
    const float* mu    = (const float*)d_in[1];   // 1024 x 4096
    const float* gamma = (const float*)d_in[2];   // scalar
    float* out = (float*)d_out;

    char* ws = (char*)d_ws;
    unsigned short* xb  = (unsigned short*)ws;                                  // 16 MB
    unsigned short* mtb = (unsigned short*)(ws + (size_t)MB * KK * 2);          //  8 MB
    float* xsq  = (float*)(ws + (size_t)MB * KK * 2 + (size_t)NN * KK * 2);     // 32 KB
    float* musq = xsq + MB;                                                     // 16 KB

    xprep <<<MB, 256, 0, stream>>>(x, xb, xsq);
    muprep<<<dim3(NN / 32, KK / 32), 256, 0, stream>>>(mu, mtb);
    musqk <<<NN / 256, 256, 0, stream>>>(mu, musq);
    rbf_gemm<<<dim3(NN / 128, MB / 128), 256, 0, stream>>>(xb, mtb, xsq, musq, gamma, out);
}
